// UpTransition_19868518711860
// MI455X (gfx1250) — matrix operand order repaired and checked by reading, not compiled, not run on hardware
//
#include <hip/hip_runtime.h>

// ---------------- problem constants (match reference) ----------------
constexpr int NC   = 10000;
constexpr int COUT = 64;
constexpr int CH   = 32;
constexpr int NF   = 4 * NC;        // 40000
constexpr int NF1  = NF + 1;        // +1 zero pad row for gather/scatter dummies
constexpr float BN_EPS = 1e-5f;

// ---------------- WMMA types (CDNA5 / gfx1250, wave32) ----------------
typedef __attribute__((ext_vector_type(16))) __bf16 v16bf;
typedef __attribute__((ext_vector_type(8)))  __bf16 v8bf;
typedef __attribute__((ext_vector_type(8)))  float  v8f;

__device__ inline v8f wmma_bf16(v16bf a, v16bf b, v8f c) {
  // D = A(16x32 bf16) * B(32x16 bf16) + C(16x16 f32)
  return __builtin_amdgcn_wmma_f32_16x16x32_bf16(
      /*neg_a=*/false, a, /*neg_b=*/false, b,
      /*c_mod=*/(short)0, c, /*reuse_a=*/false, /*reuse_b=*/false);
}

// A-matrix 16x32 bf16 fragment per ISA layout (features stored bf16 row-major):
// lane half hh: element i<8 -> K = kk + 8*hh + i ; i>=8 -> K = kk + 16 + 8*hh + (i-8)
__device__ inline v16bf load_a_frag(const __bf16* __restrict__ row, int kk, int hh) {
  v8bf x0 = *(const v8bf*)(row + kk + 8 * hh);
  v8bf x1 = *(const v8bf*)(row + kk + 16 + 8 * hh);
  return __builtin_shufflevector(x0, x1, 0, 1, 2, 3, 4, 5, 6, 7,
                                         8, 9, 10, 11, 12, 13, 14, 15);
}

__device__ inline float eluf(float x) { return x > 0.f ? x : __expf(x) - 1.f; }

// ---------------- workspace layout (float offsets) ----------------
constexpr size_t OFF_ACC1  = 0;                               // f32 [NF1,32]
constexpr size_t OFF_ACC2  = OFF_ACC1  + (size_t)NF1 * 32;    // f32 [NF1,64]
constexpr size_t OFF_ACC3  = OFF_ACC2  + (size_t)NF1 * 64;    // f32 [NF1,64]
constexpr size_t OFF_XCATF = OFF_ACC3  + (size_t)NF1 * 64;    // f32 [NF1,64] (residual)
constexpr size_t OFF_STATS = OFF_XCATF + (size_t)NF1 * 64;    // 3 * 128 floats
constexpr size_t OFF_UPB   = OFF_STATS + 384;                 // bf16 [NF1,64] (NF1*32 floats)
constexpr size_t OFF_XCATB = OFF_UPB   + (size_t)NF1 * 32;    // bf16 [NF1,64]
constexpr size_t OFF_H2B   = OFF_XCATB + (size_t)NF1 * 32;    // bf16 [NF1,64]
constexpr size_t OFF_WFRAG = OFF_H2B   + (size_t)NF1 * 32;    // bf16 weight fragments
constexpr size_t ZERO_N    = OFF_WFRAG; // zero accumulators + stats + bf16 pad rows

// bf16 fragment blocks: 512 bf16 each (32 lanes x 16), B-matrix layout per ISA:
// element i<8 -> K = 8*half + i, i>=8 -> K = 16 + 8*half + (i-8); + 32*chunk; lane&15 -> N
// f in [0,32)    : w_up      (16 N-tiles x 2 K-chunks)
// f in [32,140)  : w_upconv  (27 ko x 2 nt x 2 chunks)
// f in [140,572) : convs_w   (2 conv x 27 ko x 4 nt x 2 chunks)
constexpr int NFRAG = 572;

// m-tiles processed per wave (register-level B-fragment reuse)
constexpr int MT_CONV = 4;
constexpr int MT_UP   = 5;          // 625 m-tiles = 125 blocks * 5

// ---------------- kernels ----------------
__global__ void zero_ws_k(float* __restrict__ p, size_t n) {
  size_t t = (size_t)blockIdx.x * blockDim.x + threadIdx.x;
  size_t s = (size_t)gridDim.x * blockDim.x;
  for (; t < n; t += s) p[t] = 0.f;
}

__global__ void prep_weights_k(const float* __restrict__ w_up,
                               const float* __restrict__ w_upconv,
                               const float* __restrict__ convs_w,
                               __bf16* __restrict__ wfrag) {
  int e = blockIdx.x * blockDim.x + threadIdx.x;
  if (e >= NFRAG * 512) return;
  int f = e >> 9, s = e & 511;
  int lane = s >> 4, i = s & 15;
  int hh = lane >> 4, n = lane & 15;
  float v;
  if (f < 32) {                       // w_up as B[64 x 256]: B[c][k*64+o]=w_up[k][c][o]
    int nt = f >> 1, chunk = f & 1;
    int K = 8 * hh + (i & 7) + 16 * (i >> 3) + 32 * chunk;
    int k = nt >> 2, ob = (nt & 3) * 16;
    v = w_up[((size_t)k * 64 + K) * 64 + ob + n];
  } else if (f < 140) {               // w_upconv[27][64][32]
    int fl = f - 32;
    int ko = fl >> 2, rem = fl & 3, nt = rem >> 1, chunk = rem & 1;
    int K = 8 * hh + (i & 7) + 16 * (i >> 3) + 32 * chunk;
    v = w_upconv[((size_t)ko * 64 + K) * 32 + nt * 16 + n];
  } else {                            // convs_w[2][27][64][64]
    int fl = f - 140;
    int ic = fl / 216, r = fl % 216;
    int ko = r >> 3, rr = r & 7, nt = rr >> 1, chunk = rr & 1;
    int K = 8 * hh + (i & 7) + 16 * (i >> 3) + 32 * chunk;
    v = convs_w[(((size_t)ic * 27 + ko) * 64 + K) * 64 + nt * 16 + n];
  }
  wfrag[e] = (__bf16)v;
}

// up[n*4+k][o] = sum_c x[n][c] * w_up[k][c][o]   (A gathered from f32 x_feats)
// 128 threads (4 waves); wave ks owns k-slot ks. B frags held in regs across MT_UP tiles.
__global__ void up_gemm_k(const float* __restrict__ x,
                          const __bf16* __restrict__ wfrag,
                          __bf16* __restrict__ upb) {
  const int lane = threadIdx.x & 31;
  const int ks   = threadIdx.x >> 5;
  const int hh = lane >> 4, nl = lane & 15;
  const v16bf* bfr = (const v16bf*)wfrag;
  v16bf b[4][2];
#pragma unroll
  for (int nt = 0; nt < 4; ++nt) {
    int f = (ks * 4 + nt) * 2;
    b[nt][0] = bfr[(size_t)f * 32 + lane];
    b[nt][1] = bfr[((size_t)f + 1) * 32 + lane];
  }
#pragma unroll
  for (int mt = 0; mt < MT_UP; ++mt) {
    const int m0 = (blockIdx.x * MT_UP + mt) << 4;    // NC == 625*16 exactly
    const float* arow = x + (size_t)(m0 + nl) * 64;
    v16bf a0, a1;
#pragma unroll
    for (int i = 0; i < 8; ++i) {
      a0[i]     = (__bf16)arow[0 + 8 * hh + i];
      a0[i + 8] = (__bf16)arow[16 + 8 * hh + i];
      a1[i]     = (__bf16)arow[32 + 8 * hh + i];
      a1[i + 8] = (__bf16)arow[48 + 8 * hh + i];
    }
    v8f d[4];
#pragma unroll
    for (int nt = 0; nt < 4; ++nt) {
      v8f c = {};
      c = wmma_bf16(a0, b[nt][0], c);
      c = wmma_bf16(a1, b[nt][1], c);
      d[nt] = c;
    }
#pragma unroll
    for (int v = 0; v < 8; ++v) {
      int row = ((m0 + 8 * hh + v) << 2) + ks;
#pragma unroll
      for (int nt = 0; nt < 4; ++nt)
        upb[(size_t)row * 64 + nt * 16 + nl] = (__bf16)d[nt][v];
    }
  }
}

// sparse conv: gather(in_idx, bf16 feats) -> [16,64]x[64,16*NT] WMMA -> atomic scatter
// one wave per (MT_CONV m-tiles, kernel offset ko); B frags loaded once per wave.
template <int NT>
__global__ void sparse_conv_k(const __bf16* __restrict__ feats,   // bf16 [NF1,64]
                              const __bf16* __restrict__ wfrag,   // this conv's frags
                              float* __restrict__ acc,            // f32 [NF1, NT*16]
                              const int* __restrict__ in_idx,
                              const int* __restrict__ out_idx,
                              int mmax) {
  const int lane = threadIdx.x;             // 32 threads/block
  const int hh = lane >> 4, nl = lane & 15;
  const int ko = blockIdx.y;
  const int ib = ko * mmax;
  const int mbase = blockIdx.x * (MT_CONV * 16);

  const v16bf* bfr = (const v16bf*)wfrag;
  v16bf b[NT][2];
#pragma unroll
  for (int nt = 0; nt < NT; ++nt) {
    size_t f = (size_t)(ko * NT + nt) * 2;
    b[nt][0] = bfr[f * 32 + lane];
    b[nt][1] = bfr[(f + 1) * 32 + lane];
  }

#pragma unroll
  for (int mt = 0; mt < MT_CONV; ++mt) {
    const int m0 = mbase + mt * 16;
    int mA = m0 + nl;
    int rA = in_idx[ib + (mA < mmax ? mA : mmax - 1)];
    rA = (mA < mmax) ? rA : NF;
    const __bf16* arow = feats + (size_t)rA * 64;
    v16bf a0 = load_a_frag(arow, 0, hh);
    v16bf a1 = load_a_frag(arow, 32, hh);

    v8f d[NT];
#pragma unroll
    for (int nt = 0; nt < NT; ++nt) {
      v8f c = {};
      c = wmma_bf16(a0, b[nt][0], c);
      c = wmma_bf16(a1, b[nt][1], c);
      d[nt] = c;
    }

    int orow[8];
#pragma unroll
    for (int v = 0; v < 8; ++v) {
      int m = m0 + 8 * hh + v;
      int o = out_idx[ib + (m < mmax ? m : mmax - 1)];
      orow[v] = (m < mmax) ? o : NF;
    }
#pragma unroll
    for (int nt = 0; nt < NT; ++nt)
#pragma unroll
      for (int v = 0; v < 8; ++v)
        atomicAdd(&acc[(size_t)orow[v] * (NT * 16) + nt * 16 + nl], d[nt][v]);
  }
}

// per-channel sum / sumsq over NF rows -> stats[0:C]=sum, stats[C:2C]=sumsq
__global__ void bn_stats_k(const float* __restrict__ acc, float* __restrict__ stats, int C) {
  int t = blockIdx.x * blockDim.x + threadIdx.x;
  int c = t % C;
  int r = t / C;
  int rs = (gridDim.x * blockDim.x) / C;
  float s = 0.f, ss = 0.f;
  for (; r < NF; r += rs) {
    float v = acc[(size_t)r * C + c];
    s += v; ss += v * v;
  }
  atomicAdd(&stats[c], s);
  atomicAdd(&stats[C + c], ss);
}

// xcat[:,0:32] = ELU(BN(acc1)), xcat[:,32:64] = maxpool_t(skip)
// writes bf16 copy (conv2 input) and f32 copy (final residual)
__global__ void bn1_concat_k(const float* __restrict__ acc1, const float* __restrict__ stats,
                             const float* __restrict__ gamma, const float* __restrict__ beta,
                             const float* __restrict__ skip,
                             __bf16* __restrict__ xcatb, float* __restrict__ xcatf) {
  int t = blockIdx.x * blockDim.x + threadIdx.x;
  if (t >= NF * 64) return;
  int r = t >> 6, c = t & 63;
  float o;
  if (c < 32) {
    float mean = stats[c] * (1.f / NF);
    float var  = stats[32 + c] * (1.f / NF) - mean * mean;
    float sc = gamma[c] * rsqrtf(var + BN_EPS);
    float sh = beta[c] - mean * sc;
    o = eluf(acc1[(size_t)r * 32 + c] * sc + sh);
  } else {
    int cc = c - 32;
    o = fmaxf(skip[(size_t)(2 * r) * 32 + cc], skip[(size_t)(2 * r + 1) * 32 + cc]);
  }
  xcatf[t] = o;
  xcatb[t] = (__bf16)o;
}

// h2 = ELU(BN(acc2)) -> bf16 (conv3 input)
__global__ void bn_apply64_k(const float* __restrict__ acc, const float* __restrict__ stats,
                             const float* __restrict__ gamma, const float* __restrict__ beta,
                             __bf16* __restrict__ outb) {
  int t = blockIdx.x * blockDim.x + threadIdx.x;
  if (t >= NF * 64) return;
  int c = t & 63;
  float mean = stats[c] * (1.f / NF);
  float var  = stats[64 + c] * (1.f / NF) - mean * mean;
  float sc = gamma[c] * rsqrtf(var + BN_EPS);
  float sh = beta[c] - mean * sc;
  outb[t] = (__bf16)eluf(acc[t] * sc + sh);
}

// out = ELU( ELU(BN(acc3)) + xcat_f32 )
__global__ void final_apply_k(const float* __restrict__ acc, const float* __restrict__ stats,
                              const float* __restrict__ gamma, const float* __restrict__ beta,
                              const float* __restrict__ xcatf, float* __restrict__ out) {
  int t = blockIdx.x * blockDim.x + threadIdx.x;
  if (t >= NF * 64) return;
  int c = t & 63;
  float mean = stats[c] * (1.f / NF);
  float var  = stats[64 + c] * (1.f / NF) - mean * mean;
  float sc = gamma[c] * rsqrtf(var + BN_EPS);
  float sh = beta[c] - mean * sc;
  float h = eluf(acc[t] * sc + sh);
  out[t] = eluf(h + xcatf[t]);
}

// ---------------- host-side orchestration ----------------
extern "C" void kernel_launch(void* const* d_in, const int* in_sizes, int n_in,
                              void* d_out, int out_size, void* d_ws, size_t ws_size,
                              hipStream_t stream) {
  (void)n_in; (void)out_size; (void)ws_size;
  const float* x_feats    = (const float*)d_in[0];
  const float* skip_feats = (const float*)d_in[1];
  const float* w_up       = (const float*)d_in[2];
  const float* w_upconv   = (const float*)d_in[3];
  const float* bn1_gamma  = (const float*)d_in[4];
  const float* bn1_beta   = (const float*)d_in[5];
  const float* convs_w    = (const float*)d_in[6];
  const float* convs_g    = (const float*)d_in[7];   // [2,64]
  const float* convs_b    = (const float*)d_in[8];   // [2,64]
  const int*   in_idx     = (const int*)d_in[9];     // [27,mmax]
  const int*   out_idx    = (const int*)d_in[10];    // [27,mmax]
  const int    mmax       = in_sizes[9] / 27;

  float* ws = (float*)d_ws;
  float*  acc1  = ws + OFF_ACC1;
  float*  acc2  = ws + OFF_ACC2;
  float*  acc3  = ws + OFF_ACC3;
  float*  xcatf = ws + OFF_XCATF;
  float*  stats = ws + OFF_STATS;                    // 3 x [2*C]
  __bf16* upb   = (__bf16*)(ws + OFF_UPB);
  __bf16* xcatb = (__bf16*)(ws + OFF_XCATB);
  __bf16* h2b   = (__bf16*)(ws + OFF_H2B);
  __bf16* wfrag = (__bf16*)(ws + OFF_WFRAG);

  const int mblocks  = (mmax + 16 * MT_CONV - 1) / (16 * MT_CONV);
  const int ew_blocks = (NF * 64 + 255) / 256;

  // 1) zero accumulators / pad rows / stats (fresh every call -> deterministic)
  zero_ws_k<<<4096, 256, 0, stream>>>(ws, ZERO_N);
  // 2) weights -> bf16 B-fragment layout
  prep_weights_k<<<(NFRAG * 512 + 255) / 256, 256, 0, stream>>>(w_up, w_upconv, convs_w, wfrag);
  // 3) up-projection GEMM -> bf16 features (125 blocks * 4 waves * 5 m-tiles)
  up_gemm_k<<<NC / 16 / MT_UP, 128, 0, stream>>>(x_feats, wfrag, upb);
  // 4) conv1: [NF,64] -> [NF,32]
  sparse_conv_k<2><<<dim3(mblocks, 27), 32, 0, stream>>>(upb, wfrag + 32 * 512, acc1,
                                                         in_idx, out_idx, mmax);
  bn_stats_k<<<256, 256, 0, stream>>>(acc1, stats + 0, CH);
  bn1_concat_k<<<ew_blocks, 256, 0, stream>>>(acc1, stats + 0, bn1_gamma, bn1_beta,
                                              skip_feats, xcatb, xcatf);
  // 5) conv2: [NF,64] -> [NF,64]
  sparse_conv_k<4><<<dim3(mblocks, 27), 32, 0, stream>>>(xcatb, wfrag + 140 * 512, acc2,
                                                         in_idx, out_idx, mmax);
  bn_stats_k<<<256, 256, 0, stream>>>(acc2, stats + 128, COUT);
  bn_apply64_k<<<ew_blocks, 256, 0, stream>>>(acc2, stats + 128, convs_g, convs_b, h2b);
  // 6) conv3: [NF,64] -> [NF,64]
  sparse_conv_k<4><<<dim3(mblocks, 27), 32, 0, stream>>>(h2b, wfrag + 356 * 512, acc3,
                                                         in_idx, out_idx, mmax);
  bn_stats_k<<<256, 256, 0, stream>>>(acc3, stats + 256, COUT);
  // 7) BN + ELU + residual + ELU -> d_out
  final_apply_k<<<ew_blocks, 256, 0, stream>>>(acc3, stats + 256, convs_g + 64, convs_b + 64,
                                               xcatf, (float*)d_out);
}
